// PGCNCritic_10857677324684
// MI455X (gfx1250) — matrix operand order repaired and checked by reading, not compiled, not run on hardware
//
#include <hip/hip_runtime.h>
#include <stdint.h>

// ---------------- problem constants ----------------
#define BATCH 32
#define NN    1024
#define NODE_DIM 14
#define HID   64
#define HEAD  128

typedef __attribute__((ext_vector_type(16))) __bf16 v16bf;
typedef __attribute__((ext_vector_type(8)))  float  v8f;
typedef __attribute__((ext_vector_type(4)))  unsigned int v4u;
typedef __attribute__((ext_vector_type(8)))  int    v8i;
typedef __attribute__((ext_vector_type(4)))  int    v4i;

#if defined(__has_builtin)
# if __has_builtin(__builtin_amdgcn_tensor_load_to_lds) && __has_builtin(__builtin_amdgcn_s_wait_tensorcnt)
#  define HAVE_TDM 1
# endif
#endif
#if defined(HAVE_TDM)
# if __has_include(<hip/amd_detail/amd_gfx1250_TDM.h>)
#  define TDM_ARITY6 1
# endif
#endif

// bf16 helpers (raw bits in unsigned short)
__device__ __forceinline__ float bf2f(unsigned short s) {
    union { unsigned u; float f; } x; x.u = ((unsigned)s) << 16; return x.f;
}
__device__ __forceinline__ unsigned short f2bf(float f) {
    union { float f; unsigned u; } x; x.f = f;
    unsigned r = x.u + 0x7FFFu + ((x.u >> 16) & 1u);  // RNE
    return (unsigned short)(r >> 16);
}
__device__ __forceinline__ unsigned pack2(float a, float b) {
    return (unsigned)f2bf(a) | ((unsigned)f2bf(b) << 16);
}

// ---------------------------------------------------------------------------
// Kernel 1 (fused): one wave per adjacency row. Cache the whole 4KB row in
// VGPRs, reduce degree, then emit row-scaled bf16:  Abf[i][j] = d_i^-1/2 * A.
// Column factor d_j^-1/2 is folded into Z; row factor into the GEMM epilogue.
// Single pass over the 128MB adjacency instead of two.
// ---------------------------------------------------------------------------
__global__ void k_deg_rowscale(const float* __restrict__ adj,
                               unsigned short* __restrict__ Abf,
                               float* __restrict__ dinv) {
    int wave = threadIdx.x >> 5, lane = threadIdx.x & 31;
    int row = blockIdx.x * 8 + wave;                 // row in [0, B*N)
    const float4* r4 = (const float4*)(adj + (size_t)row * NN);
    float4 v[8];
    float s = 0.f;
#pragma unroll
    for (int i = 0; i < 8; ++i) {
        v[i] = r4[i * 32 + lane];
        s += v[i].x + v[i].y + v[i].z + v[i].w;
    }
#pragma unroll
    for (int off = 16; off > 0; off >>= 1) s += __shfl_xor(s, off);  // all lanes get total
    float di = rsqrtf(fmaxf(s, 1.0f));
    if (lane == 0) dinv[row] = di;
    uint2* o2 = (uint2*)(Abf + (size_t)row * NN);    // 256 uint2 per row
#pragma unroll
    for (int i = 0; i < 8; ++i) {
        uint2 o;
        o.x = pack2(di * v[i].x, di * v[i].y);
        o.y = pack2(di * v[i].z, di * v[i].w);
        o2[i * 32 + lane] = o;
    }
}

// ---------------------------------------------------------------------------
// Kernel 2a: Zt[b][n][k] = dinv[b,k] * sum_d x[b][k][d] * W1[d][n]
// (N-major bf16 output so WMMA B fragments are contiguous in K)
// ---------------------------------------------------------------------------
__global__ void k_transform_x(const float* __restrict__ x, const float* __restrict__ W,
                              const float* __restrict__ dinv,
                              unsigned short* __restrict__ Zt) {
    __shared__ float ws[NODE_DIM * HID];
    for (int i = threadIdx.x; i < NODE_DIM * HID; i += 256) ws[i] = W[i];
    __syncthreads();
    int kk = blockIdx.x * 256 + threadIdx.x;     // global node index
    int b = kk >> 10, k = kk & 1023;
    const float* xr = x + ((size_t)b * NN + k) * NODE_DIM;
    float acc[HID];
#pragma unroll
    for (int n = 0; n < HID; ++n) acc[n] = 0.f;
    for (int d = 0; d < NODE_DIM; ++d) {
        float xv = xr[d];
#pragma unroll
        for (int n = 0; n < HID; ++n) acc[n] += xv * ws[d * HID + n];
    }
    float dk = dinv[kk];
    for (int n = 0; n < HID; ++n)
        Zt[((size_t)b * HID + n) * NN + k] = f2bf(dk * acc[n]);
}

// ---------------------------------------------------------------------------
// Kernel 2b: Zt[b][n][k] = dinv[b,k] * sum_d Ht[b][d][k] * W[d][n]
// ---------------------------------------------------------------------------
__global__ void k_transform_h(const unsigned short* __restrict__ Ht, const float* __restrict__ W,
                              const float* __restrict__ dinv,
                              unsigned short* __restrict__ Zt) {
    __shared__ float ws[HID * HID];
    for (int i = threadIdx.x; i < HID * HID; i += 256) ws[i] = W[i];
    __syncthreads();
    int kk = blockIdx.x * 256 + threadIdx.x;
    int b = kk >> 10, k = kk & 1023;
    float acc[HID];
#pragma unroll
    for (int n = 0; n < HID; ++n) acc[n] = 0.f;
    for (int d = 0; d < HID; ++d) {
        float hv = bf2f(Ht[((size_t)b * HID + d) * NN + k]);
#pragma unroll
        for (int n = 0; n < HID; ++n) acc[n] += hv * ws[d * HID + n];
    }
    float dk = dinv[kk];
    for (int n = 0; n < HID; ++n)
        Zt[((size_t)b * HID + n) * NN + k] = f2bf(dk * acc[n]);
}

// ---------------------------------------------------------------------------
// K-chunk geometry for the WMMA GEMM: 128-wide K chunks, 64 rows of Zt,
// LDS row stride 272B (256B data + 16B pad -> conflict-free ds_load_b128).
// ---------------------------------------------------------------------------
#define KCHUNK   128
#define NCHUNKS  (NN / KCHUNK)           // 8
#define LDSROW4  17                      // uint4 per padded LDS row (272B)

#if defined(HAVE_TDM)
__device__ __forceinline__ void tdm_load_tile(const unsigned short* gsrc, unsigned lds_off) {
    unsigned long long ga = (unsigned long long)(size_t)gsrc;
    v4u g0 = { 1u,                                   // count=1 (valid), user mode
               lds_off,
               (unsigned)(ga & 0xFFFFFFFFu),
               (unsigned)((ga >> 32) & 0x01FFFFFFu) | 0x80000000u };   // type=2
    // data_size=2B, pad_enable, pad every 64 DWORDs (256B) by 4 DWORDs (16B),
    // tensor 1024x64 (row stride 1024), tile 128x64
    v8i g1 = { (int)((1u << 16) | (1u << 20) | (5u << 22) | (3u << 25)),
               (int)(1024u << 16),        // tensor_dim0 = 1024  (bits 79:48)
               (int)(64u << 16),          // tensor_dim1 = 64    (bits 111:80)
               (int)((unsigned)KCHUNK << 16), // tile_dim0 = 128 (bits 127:112)
               (int)(64u),                // tile_dim1 = 64
               (int)(1024u),              // tensor_dim0_stride = 1024
               0, 0 };
    v4i g2 = { 0, 0, 0, 0 };
    v4i g3 = { 0, 0, 0, 0 };
# if defined(TDM_ARITY6)
    v8i g4 = { 0, 0, 0, 0, 0, 0, 0, 0 };
    __builtin_amdgcn_tensor_load_to_lds(g0, g1, g2, g3, g4, 0);
# else
    __builtin_amdgcn_tensor_load_to_lds(g0, g1, g2, g3, 0);
# endif
}
#endif

// ---------------------------------------------------------------------------
// Kernel 3: H = relu(d_m^-1/2 * (A_rowscaled @ Z') + bias)
// Grid: (8 m-tiles, 32 batches), 256 threads = 8 waves. Wave w owns rows
// mtile*128 + w*16 .. +15 and all 64 features (4 wmma accumulators).
// Double-buffered LDS tiles filled by the Tensor Data Mover, overlapped
// with WMMA compute. Output transposed: Ht[b][f][node] (bf16).
// ---------------------------------------------------------------------------
__global__ void __launch_bounds__(256)
k_gcn_wmma(const unsigned short* __restrict__ Abf, const unsigned short* __restrict__ Zt,
           const float* __restrict__ bias, const float* __restrict__ dinv,
           unsigned short* __restrict__ Ht) {
    __shared__ uint4 zs4[2][64 * LDSROW4];           // 2 x 17408 B
    const int b = blockIdx.y, mtile = blockIdx.x;
    const int tid = threadIdx.x, wave = tid >> 5, lane = tid & 31;
    const int laneLo = lane & 15, laneHi = lane >> 4;

    const int m = mtile * 128 + wave * 16 + laneLo;
    const uint4* arow4 = (const uint4*)(Abf + (((size_t)b * NN) + m) * NN);
    const unsigned short* zbase = Zt + ((size_t)b * HID) * NN;

    v8f acc[4];
#pragma unroll
    for (int nt = 0; nt < 4; ++nt)
#pragma unroll
        for (int r = 0; r < 8; ++r) acc[nt][r] = 0.f;

    // ---- prologue: fill buffer 0 with chunk 0 ----
#if defined(HAVE_TDM)
    if (wave == 0) {
        tdm_load_tile(zbase, (unsigned)(size_t)(&zs4[0][0]));
        __builtin_amdgcn_s_wait_tensorcnt(0);
    }
#else
    for (int i = tid; i < 64 * (KCHUNK / 8); i += 256) {       // 1024 uint4
        int row = i >> 4, c = i & 15;
        zs4[0][row * LDSROW4 + c] = ((const uint4*)(zbase + (size_t)row * NN))[c];
    }
#endif
    __syncthreads();

    int buf = 0;
#if !defined(HAVE_TDM)
    uint4 pf[4];
#endif
    for (int chunk = 0; chunk < NCHUNKS; ++chunk) {
        // ---- kick off next chunk's fill into the other buffer ----
#if defined(HAVE_TDM)
        if (chunk < NCHUNKS - 1 && wave == 0)
            tdm_load_tile(zbase + (chunk + 1) * KCHUNK,
                          (unsigned)(size_t)(&zs4[buf ^ 1][0]));
#else
        if (chunk < NCHUNKS - 1) {
#pragma unroll
            for (int j = 0; j < 4; ++j) {
                int i = tid + j * 256;                   // 1024 uint4 staged in regs
                int row = i >> 4, c = i & 15;
                pf[j] = ((const uint4*)(zbase + (size_t)row * NN + (chunk + 1) * KCHUNK))[c];
            }
        }
#endif
        // ---- compute on current buffer ----
#pragma unroll
        for (int ks = 0; ks < KCHUNK / 32; ++ks) {       // 4 k-steps of 32
            const int kq = chunk * (KCHUNK / 8) + ks * 4 + laneHi;
            union { uint4 q[2]; v16bf v; } af;
            af.q[0] = arow4[kq];
            af.q[1] = arow4[kq + 2];
#pragma unroll
            for (int nt = 0; nt < 4; ++nt) {
                const int zb = (nt * 16 + laneLo) * LDSROW4 + (ks << 2) + laneHi;
                union { uint4 q[2]; v16bf v; } bfr;
                bfr.q[0] = zs4[buf][zb];
                bfr.q[1] = zs4[buf][zb + 2];
                acc[nt] = __builtin_amdgcn_wmma_f32_16x16x32_bf16(
                    false, af.v, false, bfr.v, (short)0, acc[nt], false, false);
            }
        }
        // ---- publish next buffer ----
        if (chunk < NCHUNKS - 1) {
            __syncthreads();                             // everyone done with zs4[buf]
#if defined(HAVE_TDM)
            if (wave == 0) __builtin_amdgcn_s_wait_tensorcnt(0);
#else
#pragma unroll
            for (int j = 0; j < 4; ++j) {
                int i = tid + j * 256;
                int row = i >> 4, c = i & 15;
                zs4[buf ^ 1][row * LDSROW4 + c] = pf[j];
            }
#endif
            __syncthreads();                             // next tile visible to all
        }
        buf ^= 1;
    }

    // ---- epilogue: d_m^-1/2 scaling + bias + relu, store Ht[b][f][node] ----
    const int nodeBase = mtile * 128 + wave * 16 + laneHi * 8;
    const float4 d0 = *(const float4*)(dinv + (size_t)b * NN + nodeBase);
    const float4 d1 = *(const float4*)(dinv + (size_t)b * NN + nodeBase + 4);
    const float dm[8] = { d0.x, d0.y, d0.z, d0.w, d1.x, d1.y, d1.z, d1.w };
#pragma unroll
    for (int nt = 0; nt < 4; ++nt) {
        const int f = nt * 16 + laneLo;
        const float bv = bias[f];
        float r[8];
#pragma unroll
        for (int i = 0; i < 8; ++i) r[i] = fmaxf(dm[i] * acc[nt][i] + bv, 0.f);
        uint4 o;
        o.x = pack2(r[0], r[1]);
        o.y = pack2(r[2], r[3]);
        o.z = pack2(r[4], r[5]);
        o.w = pack2(r[6], r[7]);
        size_t dst = ((size_t)b * HID + f) * NN + nodeBase;
        *(uint4*)(Ht + dst) = o;
    }
}

// ---------------------------------------------------------------------------
// Kernel 4: pooled[b][f] = mean_k Ht[b][f][k].  One wave per (b,f).
// ---------------------------------------------------------------------------
__global__ void k_pool(const unsigned short* __restrict__ Ht, float* __restrict__ pooled) {
    int g = blockIdx.x * 8 + (threadIdx.x >> 5);   // (b,f) pair, g < 2048
    int lane = threadIdx.x & 31;
    const uint2* p = (const uint2*)(Ht + (size_t)g * NN);
    float s = 0.f;
#pragma unroll
    for (int i = 0; i < 8; ++i) {
        uint2 v = p[i * 32 + lane];
        s += bf2f((unsigned short)(v.x & 0xFFFF)) + bf2f((unsigned short)(v.x >> 16));
        s += bf2f((unsigned short)(v.y & 0xFFFF)) + bf2f((unsigned short)(v.y >> 16));
    }
#pragma unroll
    for (int off = 16; off > 0; off >>= 1) s += __shfl_xor(s, off);
    if (lane == 0) pooled[g] = s * (1.0f / (float)NN);
}

// ---------------------------------------------------------------------------
// Kernel 5: head MLP: out[b] = relu(pooled@Wf1+bf1) @ Wf2 + bf2
// One block (128 threads) per batch element.
// ---------------------------------------------------------------------------
__global__ void k_head(const float* __restrict__ pooled,
                       const float* __restrict__ Wf1, const float* __restrict__ bf1,
                       const float* __restrict__ Wf2, const float* __restrict__ bf2,
                       float* __restrict__ out) {
    __shared__ float red[HEAD];
    int b = blockIdx.x, t = threadIdx.x;
    const float* pr = pooled + (size_t)b * HID;
    float s = 0.f;
#pragma unroll
    for (int d = 0; d < HID; ++d) s += pr[d] * Wf1[d * HEAD + t];
    float h = fmaxf(s + bf1[t], 0.f);
    red[t] = h * Wf2[t];
    __syncthreads();
    for (int stride = HEAD / 2; stride > 0; stride >>= 1) {
        if (t < stride) red[t] += red[t + stride];
        __syncthreads();
    }
    if (t == 0) out[b] = red[0] + bf2[0];
}

// ---------------------------------------------------------------------------
extern "C" void kernel_launch(void* const* d_in, const int* in_sizes, int n_in,
                              void* d_out, int out_size, void* d_ws, size_t ws_size,
                              hipStream_t stream) {
    (void)in_sizes; (void)n_in; (void)out_size; (void)ws_size;
    const float* x   = (const float*)d_in[0];
    const float* adj = (const float*)d_in[1];
    const float* W1  = (const float*)d_in[2];
    const float* b1  = (const float*)d_in[3];
    const float* W2  = (const float*)d_in[4];
    const float* b2  = (const float*)d_in[5];
    const float* W3  = (const float*)d_in[6];
    const float* b3  = (const float*)d_in[7];
    const float* Wf1 = (const float*)d_in[8];
    const float* bf1 = (const float*)d_in[9];
    const float* Wf2 = (const float*)d_in[10];
    const float* bf2 = (const float*)d_in[11];
    float* out = (float*)d_out;

    char* ws = (char*)d_ws;
    unsigned short* Abf = (unsigned short*)ws;                           // 64 MB (row-scaled bf16 A)
    unsigned short* Zt  = (unsigned short*)(ws + ((size_t)64 << 20));    // 4 MB
    unsigned short* Ht  = (unsigned short*)(ws + ((size_t)68 << 20));    // 4 MB
    float* dinv   = (float*)(ws + ((size_t)72 << 20));                   // 128 KB
    float* pooled = (float*)(ws + ((size_t)72 << 20) + (131072));        // 8 KB

    dim3 gemmGrid(8, BATCH);

    // single pass over adjacency: degree + row-scale + bf16 convert
    k_deg_rowscale<<<4096, 256, 0, stream>>>(adj, Abf, dinv);

    // layer 1
    k_transform_x<<<128, 256, 0, stream>>>(x, W1, dinv, Zt);
    k_gcn_wmma   <<<gemmGrid, 256, 0, stream>>>(Abf, Zt, b1, dinv, Ht);
    // layer 2
    k_transform_h<<<128, 256, 0, stream>>>(Ht, W2, dinv, Zt);
    k_gcn_wmma   <<<gemmGrid, 256, 0, stream>>>(Abf, Zt, b2, dinv, Ht);
    // layer 3
    k_transform_h<<<128, 256, 0, stream>>>(Ht, W3, dinv, Zt);
    k_gcn_wmma   <<<gemmGrid, 256, 0, stream>>>(Abf, Zt, b3, dinv, Ht);

    // mean pool + head MLP
    k_pool<<<256, 256, 0, stream>>>(Ht, pooled);
    k_head<<<BATCH, HEAD, 0, stream>>>(pooled, Wf1, bf1, Wf2, bf2, out);
}
